// GAT_GCNNet_30786325578056
// MI455X (gfx1250) — matrix operand order repaired and checked by reading, not compiled, not run on hardware
//
#include <hip/hip_runtime.h>
#include <math.h>

typedef __attribute__((ext_vector_type(16))) _Float16 v16h;
typedef __attribute__((ext_vector_type(8)))  _Float16 v8h;
typedef __attribute__((ext_vector_type(8)))  float    v8f;
typedef __attribute__((ext_vector_type(4)))  unsigned v4u;
typedef __attribute__((ext_vector_type(8)))  int      v8i;
typedef __attribute__((ext_vector_type(4)))  int      v4i;

#define NG 512   // NUM_GRAPHS

#if defined(__gfx1250__) && __has_builtin(__builtin_amdgcn_tensor_load_to_lds)
#define USE_TDM 1
#if __has_include(<hip/amd_detail/amd_gfx1250_TDM.h>)
#define TDM_SIX_ARGS 1
#endif
#endif

// ---------------------------------------------------------------- helpers
__device__ __forceinline__ float leaky02(float x) { return x > 0.f ? x : 0.2f * x; }

__device__ __forceinline__ void atomicMaxF(float* addr, float val) {
    int* ai = (int*)addr;
    int cur = __float_as_int(*addr);
    while (__int_as_float(cur) < val) {
        int old = atomicCAS(ai, cur, __float_as_int(val));
        if (old == cur) break;
        cur = old;
    }
}

__global__ void fill_f32(float* p, int n, float v) {
    int i = blockIdx.x * blockDim.x + threadIdx.x;
    if (i < n) p[i] = v;
}

// ---------------------------------------------------------------- packing
// A[M,K] f32 -> Ah[M,Kp] f16 (K zero-padded to multiple of 32)
__global__ void pack_a(const float* __restrict__ A, _Float16* __restrict__ Ah,
                       int M, int K, int Kp) {
    long long i = (long long)blockIdx.x * blockDim.x + threadIdx.x;
    if (i >= (long long)M * Kp) return;
    int r = (int)(i / Kp), k = (int)(i % Kp);
    Ah[i] = (k < K) ? (_Float16)A[(size_t)r * K + k] : (_Float16)0.f;
}
// W[K,N] f32 -> Wt[Np,Kp] f16 transposed, zero-padded (Np mult of 64)
__global__ void pack_wt(const float* __restrict__ W, _Float16* __restrict__ Wt,
                        int K, int N, int Kp, int Np) {
    long long i = (long long)blockIdx.x * blockDim.x + threadIdx.x;
    if (i >= (long long)Np * Kp) return;
    int n = (int)(i / Kp), k = (int)(i % Kp);
    Wt[i] = (k < K && n < N) ? (_Float16)W[(size_t)k * N + n] : (_Float16)0.f;
}

// ---------------------------------------------------------------- TDM issue
#ifdef USE_TDM
// 2D tile load: rows x 32 f16 elements, row stride kp elements, into LDS @ lds_off.
// D# per CDNA5 ISA ch.8: group0 = {flags, lds_addr, global_addr[31:0],
// global_addr[56:32]|type<<30}; group1 packs data_size/dims/tiles/strides.
__device__ __forceinline__ void tdm_load_2d(unsigned lds_off, const _Float16* gsrc,
                                            unsigned rows, unsigned kp) {
    unsigned long long ga = (unsigned long long)(uintptr_t)gsrc;
    v4u g0;
    g0[0] = 1u;                                   // count=1, user descriptor
    g0[1] = lds_off;                              // lds_addr (bytes)
    g0[2] = (unsigned)ga;                         // global_addr[31:0]
    g0[3] = (unsigned)((ga >> 32) & 0x01FFFFFFu) | (2u << 30);  // addr[56:32], type=2
    v8i g1;
    g1[0] = 0x10000;                              // workgroup_mask=0, data_size=1 (2B)
    g1[1] = (int)((kp & 0xffffu) << 16);          // tensor_dim0[15:0]
    g1[2] = (int)((kp >> 16) | (rows << 16));     // tensor_dim0[31:16], tensor_dim1[15:0]
    g1[3] = (int)(32u << 16);                     // tensor_dim1[31:16]=0, tile_dim0=32
    g1[4] = (int)rows;                            // tile_dim1=rows, tile_dim2=0
    g1[5] = (int)kp;                              // tensor_dim0_stride[31:0]
    g1[6] = 0;                                    // stride0[47:32]=0, stride1[15:0]=0
    g1[7] = 0;
    v4i z4; z4[0] = z4[1] = z4[2] = z4[3] = 0;    // groups 2/3 unused (2D)
#ifdef TDM_SIX_ARGS
    v8i z8; for (int q = 0; q < 8; ++q) z8[q] = 0;
    __builtin_amdgcn_tensor_load_to_lds(g0, g1, z4, z4, z8, 0);
#else
    __builtin_amdgcn_tensor_load_to_lds(g0, g1, z4, z4, 0);
#endif
}
#endif

// ---------------------------------------------------------------- WMMA GEMM
// C[m, colOff+n] = act( Ah[M,Kp] @ Wt[Np,Kp]^T + bias[n] )   (f16 in, f32 out)
// 128 threads = 4 waves; block covers 16 rows x 64 cols; wave w -> cols w*16..+15.
// K-tiles of 32 staged to LDS via Tensor Data Mover (fallback: b128 copies).
__global__ void gemm_wmma_f16(const _Float16* __restrict__ Ah, const _Float16* __restrict__ Wt,
                              const float* __restrict__ bias, float* __restrict__ C,
                              int Kp, int N, int ldc, int colOff, int act) {
    __shared__ __align__(16) _Float16 lA[16 * 32];
    __shared__ __align__(16) _Float16 lB[64 * 32];

    const int tid   = threadIdx.x;
    const int wv    = tid >> 5;
    const int lane  = tid & 31;
    const int mBase = blockIdx.y * 16;
    const int nB0   = blockIdx.x * 64;

    v8f acc = {};
    for (int k0 = 0; k0 < Kp; k0 += 32) {
#ifdef USE_TDM
        if (tid == 0) {
            tdm_load_2d((unsigned)(uintptr_t)&lA[0], Ah + (size_t)mBase * Kp + k0, 16u, (unsigned)Kp);
            tdm_load_2d((unsigned)(uintptr_t)&lB[0], Wt + (size_t)nB0  * Kp + k0, 64u, (unsigned)Kp);
        }
        if (wv == 0) __builtin_amdgcn_s_wait_tensorcnt(0);
        __syncthreads();
#else
        for (int i = tid; i < 64; i += 128) {         // A: 16 rows x 4 chunks of 16B
            int r = i >> 2, s = i & 3;
            *(v8h*)&lA[r * 32 + s * 8] = *(const v8h*)&Ah[(size_t)(mBase + r) * Kp + k0 + s * 8];
        }
        for (int i = tid; i < 256; i += 128) {        // B: 64 rows x 4 chunks of 16B
            int r = i >> 2, s = i & 3;
            *(v8h*)&lB[r * 32 + s * 8] = *(const v8h*)&Wt[(size_t)(nB0 + r) * Kp + k0 + s * 8];
        }
        __syncthreads();
#endif
        // A frag: lane row = lane&15; halves K = 8*hi..+7 and 16+8*hi..+7
        const int hi = lane >> 4;
        const int rw = lane & 15;
        v8h alo = *(const v8h*)&lA[rw * 32 + hi * 8];
        v8h ahi = *(const v8h*)&lA[rw * 32 + 16 + hi * 8];
        v16h af;
#pragma unroll
        for (int i = 0; i < 8; ++i) { af[i] = alo[i]; af[8 + i] = ahi[i]; }
        // B frag: lane col = lane&15; halves K = 8*hi..+7 and 16+8*hi..+7
        const _Float16* pb = &lB[(wv * 16 + rw) * 32 + hi * 8];
        v8h blo = *(const v8h*)pb;
        v8h bhi = *(const v8h*)(pb + 16);
        v16h bf;
#pragma unroll
        for (int i = 0; i < 8; ++i) { bf[i] = blo[i]; bf[8 + i] = bhi[i]; }

        acc = __builtin_amdgcn_wmma_f32_16x16x32_f16(false, af, false, bf,
                                                     (short)0, acc, false, false);
        __syncthreads();
    }
    // C/D layout: lane col = lane&15; VGPR r -> row r + 8*(lane>>4)
    const int hi = lane >> 4;
    const int gc = nB0 + wv * 16 + (lane & 15);
    if (gc < N) {
        float bv = bias ? bias[gc] : 0.f;
#pragma unroll
        for (int r = 0; r < 8; ++r) {
            int gr = mBase + r + 8 * hi;
            float v = acc[r] + bv;
            if (act == 1)      v = fmaxf(v, 0.f);
            else if (act == 2) v = v > 0.f ? v : (expf(v) - 1.f);
            C[(size_t)gr * ldc + colOff + gc] = v;
        }
    }
}

// ---------------------------------------------------------------- GAT pieces
__global__ void gat_attn(const float* __restrict__ h, const float* __restrict__ asrc,
                         const float* __restrict__ adst, float* __restrict__ al,
                         float* __restrict__ ar, int n, int H, int Cc) {
    int i = blockIdx.x * blockDim.x + threadIdx.x;
    if (i >= n * H) return;
    int node = i / H, hd = i % H;
    const float* hp = h + (size_t)node * H * Cc + (size_t)hd * Cc;
    const float* sa = asrc + hd * Cc;
    const float* da = adst + hd * Cc;
    float s0 = 0.f, s1 = 0.f;
    for (int c = 0; c < Cc; ++c) { float v = hp[c]; s0 += v * sa[c]; s1 += v * da[c]; }
    al[i] = s0; ar[i] = s1;
}

// virtual edges: e < E -> (src[e],dst[e]); e >= E -> self loop (e-E, e-E)
__global__ void gat_edge_max(const int* __restrict__ src, const int* __restrict__ dst,
                             int E, int Nn, const float* __restrict__ al,
                             const float* __restrict__ ar, float* __restrict__ m, int H) {
    int i = blockIdx.x * blockDim.x + threadIdx.x;
    if (i >= (E + Nn) * H) return;
    int e = i / H, hd = i % H;
    int s, d;
    if (e < E) { s = src[e]; d = dst[e]; } else { s = d = e - E; }
    float v = leaky02(al[s * H + hd] + ar[d * H + hd]);
    atomicMaxF(&m[d * H + hd], v);
}

__global__ void gat_edge_sum(const int* __restrict__ src, const int* __restrict__ dst,
                             int E, int Nn, const float* __restrict__ al,
                             const float* __restrict__ ar, const float* __restrict__ m,
                             float* __restrict__ ssum, int H) {
    int i = blockIdx.x * blockDim.x + threadIdx.x;
    if (i >= (E + Nn) * H) return;
    int e = i / H, hd = i % H;
    int s, d;
    if (e < E) { s = src[e]; d = dst[e]; } else { s = d = e - E; }
    float v = leaky02(al[s * H + hd] + ar[d * H + hd]);
    atomicAdd(&ssum[d * H + hd], expf(v - m[d * H + hd]));
}

__global__ void gat_edge_scatter(const int* __restrict__ src, const int* __restrict__ dst,
                                 int E, int Nn, const float* __restrict__ al,
                                 const float* __restrict__ ar, const float* __restrict__ m,
                                 const float* __restrict__ ssum, const float* __restrict__ h,
                                 float* __restrict__ out, int H, int Cc) {
    int i = blockIdx.x * blockDim.x + threadIdx.x;
    if (i >= (E + Nn) * H) return;
    int e = i / H, hd = i % H;
    int s, d;
    if (e < E) { s = src[e]; d = dst[e]; } else { s = d = e - E; }
    float v = leaky02(al[s * H + hd] + ar[d * H + hd]);
    float alpha = expf(v - m[d * H + hd]) / (ssum[d * H + hd] + 1e-16f);
    const float* hp = h + (size_t)s * H * Cc + (size_t)hd * Cc;
    float* op = out + (size_t)d * H * Cc + (size_t)hd * Cc;
    for (int c = 0; c < Cc; ++c) atomicAdd(&op[c], hp[c] * alpha);
}

__global__ void bias_act_ip(float* __restrict__ p, const float* __restrict__ b,
                            int n, int F, int act) {
    int i = blockIdx.x * blockDim.x + threadIdx.x;
    if (i >= n * F) return;
    float v = p[i] + b[i % F];
    if (act == 1)      v = fmaxf(v, 0.f);
    else if (act == 2) v = v > 0.f ? v : (expf(v) - 1.f);
    p[i] = v;
}

// ---------------------------------------------------------------- GCN pieces
__global__ void gcn_deg(const int* __restrict__ dst, int E, float* __restrict__ deg) {
    int i = blockIdx.x * blockDim.x + threadIdx.x;
    if (i < E) atomicAdd(&deg[dst[i]], 1.f);
}

__global__ void rsqrt_ip(float* __restrict__ p, int n) {
    int i = blockIdx.x * blockDim.x + threadIdx.x;
    if (i < n) p[i] = rsqrtf(fmaxf(p[i], 1.f));
}

__global__ void gcn_scatter(const int* __restrict__ src, const int* __restrict__ dst,
                            int E, int Nn, const float* __restrict__ dis,
                            const float* __restrict__ h, float* __restrict__ out, int F) {
    int e = blockIdx.x * blockDim.x + threadIdx.x;
    if (e >= E + Nn) return;
    int s, d;
    if (e < E) { s = src[e]; d = dst[e]; } else { s = d = e - E; }
    float nm = dis[s] * dis[d];
    const float* hp = h + (size_t)s * F;
    float* op = out + (size_t)d * F;
    for (int f = 0; f < F; ++f) atomicAdd(&op[f], hp[f] * nm);
}

// ---------------------------------------------------------------- pooling / head
__global__ void seg_max(const float* __restrict__ x, const int* __restrict__ batch,
                        float* __restrict__ out, int n, int F) {
    int i = blockIdx.x * blockDim.x + threadIdx.x;
    if (i >= n * F) return;
    int node = i / F, f = i % F;
    atomicMaxF(&out[(size_t)batch[node] * F + f], x[i]);
}

__global__ void head_out(const float* __restrict__ x, const float* __restrict__ w,
                         const float* __restrict__ b, float* __restrict__ out, int M, int K) {
    int i = blockIdx.x * blockDim.x + threadIdx.x;
    if (i >= M) return;
    float s = 0.f;
    for (int k = 0; k < K; ++k) s += x[(size_t)i * K + k] * w[k];
    out[i] = s + b[0];
}

// ---------------------------------------------------------------- launch
static inline int ce(long long a, long long b) { return (int)((a + b - 1) / b); }

extern "C" void kernel_launch(void* const* d_in, const int* in_sizes, int n_in,
                              void* d_out, int out_size, void* d_ws, size_t ws_size,
                              hipStream_t stream) {
    (void)n_in; (void)out_size; (void)ws_size;
    const float* x_drug = (const float*)d_in[0];
    const int*   ei_d   = (const int*)d_in[1];
    const int*   bat_d  = (const int*)d_in[2];
    const float* x_prot = (const float*)d_in[3];
    const int*   ei_p   = (const int*)d_in[4];
    const int*   bat_p  = (const int*)d_in[5];
    const float* W1     = (const float*)d_in[6];
    const float* as1    = (const float*)d_in[7];
    const float* ad1    = (const float*)d_in[8];
    const float* b1     = (const float*)d_in[9];
    const float* W2     = (const float*)d_in[10];
    const float* as2    = (const float*)d_in[11];
    const float* ad2    = (const float*)d_in[12];
    const float* b2     = (const float*)d_in[13];
    const float* fcg1W  = (const float*)d_in[14];
    const float* fcg1b  = (const float*)d_in[15];
    const float* g1W    = (const float*)d_in[16];
    const float* g1b    = (const float*)d_in[17];
    const float* g2W    = (const float*)d_in[18];
    const float* g2b    = (const float*)d_in[19];
    const float* l1W    = (const float*)d_in[20];
    const float* l1b    = (const float*)d_in[21];
    const float* fc1W   = (const float*)d_in[22];
    const float* fc1b   = (const float*)d_in[23];
    const float* fc2W   = (const float*)d_in[24];
    const float* fc2b   = (const float*)d_in[25];
    const float* outW   = (const float*)d_in[26];
    const float* outb   = (const float*)d_in[27];
    float* outp = (float*)d_out;

    const int nd = in_sizes[0] / 78;
    const int ed = in_sizes[1] / 2;
    const int np = in_sizes[3] / 41;
    const int ep = in_sizes[4] / 2;
    const int* src_d = ei_d;           const int* dst_d = ei_d + ed;
    const int* src_p = ei_p;           const int* dst_p = ei_p + ep;

    // ---- workspace bump allocator (float units), with region reuse
    float* WS = (float*)d_ws;
    size_t off = 0;
    auto alloc = [&](size_t n) { size_t o = off; off += (n + 63) & ~(size_t)63; return o; };
    const size_t rA  = alloc((size_t)nd * 780);   // h1 ; later h2 (nd*128) + acc2 (nd*128)
    const size_t rB  = alloc((size_t)nd * 780);   // acc1/o1 ; later protein buffers
    const size_t al1 = alloc((size_t)nd * 10), ar1 = alloc((size_t)nd * 10);
    const size_t m1  = alloc((size_t)nd * 10), s1  = alloc((size_t)nd * 10);
    const size_t al2 = alloc(nd), ar2 = alloc(nd), m2 = alloc(nd), s2 = alloc(nd);
    const size_t pd  = alloc((size_t)NG * 128);
    const size_t pp  = alloc((size_t)NG * 64);
    const size_t xc  = alloc((size_t)NG * 256);
    const size_t t1  = alloc((size_t)NG * 1024);
    const size_t t2  = alloc((size_t)NG * 256);
    const size_t deg = alloc(np);
    const size_t aF  = alloc((size_t)nd * 400);   // f16 packed A (nd*800 halves max)
    const size_t wF  = alloc(140000);             // f16 packed W^T (<=262144 halves)
    // region overlays
    float* h1  = WS + rA;
    float* o1  = WS + rB;                      // acc1 then elu'd in place
    float* h2  = WS + rA;                      // overwrites h1 (dead)
    float* ac2 = WS + rA + (size_t)nd * 128;   // GAT2 accumulation
    float* hp1 = WS + rB;                      // protein (o1 dead after h2 gemm)
    float* ap1 = WS + rB + (size_t)np * 41;
    float* hp2 = WS + rB + (size_t)np * 82;
    float* ap2 = WS + rB + (size_t)np * 146;
    _Float16* Ah16 = (_Float16*)(WS + aF);
    _Float16* Wt16 = (_Float16*)(WS + wF);

    const int T = 256;
    const float NEG = -3.402823466e+38f;
    auto gemm = [&](const float* A, const float* Wm, const float* bias, float* C,
                    int M, int K, int N, int ldc, int colOff, int act) {
        int Kp = ((K + 31) / 32) * 32;
        int Np = ((N + 63) / 64) * 64;
        pack_a<<<ce((long long)M * Kp, T), T, 0, stream>>>(A, Ah16, M, K, Kp);
        pack_wt<<<ce((long long)Np * Kp, T), T, 0, stream>>>(Wm, Wt16, K, N, Kp, Np);
        dim3 g(Np / 64, M / 16);
        gemm_wmma_f16<<<g, 128, 0, stream>>>(Ah16, Wt16, bias, C, Kp, N, ldc, colOff, act);
    };

    // ================= drug branch =================
    // GAT1: 78 -> 780 (H=10, C=78)
    gemm(x_drug, W1, nullptr, h1, nd, 78, 780, 780, 0, 0);
    gat_attn<<<ce((long long)nd * 10, T), T, 0, stream>>>(h1, as1, ad1, WS + al1, WS + ar1, nd, 10, 78);
    fill_f32<<<ce((long long)nd * 780, T), T, 0, stream>>>(o1, nd * 780, 0.f);
    fill_f32<<<ce((long long)nd * 10, T), T, 0, stream>>>(WS + m1, nd * 10, NEG);
    fill_f32<<<ce((long long)nd * 10, T), T, 0, stream>>>(WS + s1, nd * 10, 0.f);
    {
        long long tot = (long long)(ed + nd) * 10;
        gat_edge_max<<<ce(tot, T), T, 0, stream>>>(src_d, dst_d, ed, nd, WS + al1, WS + ar1, WS + m1, 10);
        gat_edge_sum<<<ce(tot, T), T, 0, stream>>>(src_d, dst_d, ed, nd, WS + al1, WS + ar1, WS + m1, WS + s1, 10);
        gat_edge_scatter<<<ce(tot, T), T, 0, stream>>>(src_d, dst_d, ed, nd, WS + al1, WS + ar1,
                                                       WS + m1, WS + s1, h1, o1, 10, 78);
    }
    bias_act_ip<<<ce((long long)nd * 780, T), T, 0, stream>>>(o1, b1, nd, 780, 2);  // +b1, elu

    // GAT2: 780 -> 128 (H=1)
    gemm(o1, W2, nullptr, h2, nd, 780, 128, 128, 0, 0);
    gat_attn<<<ce(nd, T), T, 0, stream>>>(h2, as2, ad2, WS + al2, WS + ar2, nd, 1, 128);
    fill_f32<<<ce((long long)nd * 128, T), T, 0, stream>>>(ac2, nd * 128, 0.f);
    fill_f32<<<ce(nd, T), T, 0, stream>>>(WS + m2, nd, NEG);
    fill_f32<<<ce(nd, T), T, 0, stream>>>(WS + s2, nd, 0.f);
    {
        long long tot = (long long)(ed + nd);
        gat_edge_max<<<ce(tot, T), T, 0, stream>>>(src_d, dst_d, ed, nd, WS + al2, WS + ar2, WS + m2, 1);
        gat_edge_sum<<<ce(tot, T), T, 0, stream>>>(src_d, dst_d, ed, nd, WS + al2, WS + ar2, WS + m2, WS + s2, 1);
        gat_edge_scatter<<<ce(tot, T), T, 0, stream>>>(src_d, dst_d, ed, nd, WS + al2, WS + ar2,
                                                       WS + m2, WS + s2, h2, ac2, 1, 128);
    }
    bias_act_ip<<<ce((long long)nd * 128, T), T, 0, stream>>>(ac2, b2, nd, 128, 1);  // +b2, relu
    // gmp + fc_g1 -> xc[:, :128]
    fill_f32<<<ce(NG * 128, T), T, 0, stream>>>(WS + pd, NG * 128, NEG);
    seg_max<<<ce((long long)nd * 128, T), T, 0, stream>>>(ac2, bat_d, WS + pd, nd, 128);
    gemm(WS + pd, fcg1W, fcg1b, WS + xc, NG, 128, 128, 256, 0, 1);

    // ================= protein branch =================
    fill_f32<<<ce(np, T), T, 0, stream>>>(WS + deg, np, 1.f);   // self loop contributes 1
    gcn_deg<<<ce(ep, T), T, 0, stream>>>(dst_p, ep, WS + deg);
    rsqrt_ip<<<ce(np, T), T, 0, stream>>>(WS + deg, np);        // deg -> dis in place
    // GCN1: 41 -> 41
    gemm(x_prot, g1W, nullptr, hp1, np, 41, 41, 41, 0, 0);
    fill_f32<<<ce((long long)np * 41, T), T, 0, stream>>>(ap1, np * 41, 0.f);
    gcn_scatter<<<ce(ep + np, T), T, 0, stream>>>(src_p, dst_p, ep, np, WS + deg, hp1, ap1, 41);
    bias_act_ip<<<ce((long long)np * 41, T), T, 0, stream>>>(ap1, g1b, np, 41, 1);
    // GCN2: 41 -> 64
    gemm(ap1, g2W, nullptr, hp2, np, 41, 64, 64, 0, 0);
    fill_f32<<<ce((long long)np * 64, T), T, 0, stream>>>(ap2, np * 64, 0.f);
    gcn_scatter<<<ce(ep + np, T), T, 0, stream>>>(src_p, dst_p, ep, np, WS + deg, hp2, ap2, 64);
    bias_act_ip<<<ce((long long)np * 64, T), T, 0, stream>>>(ap2, g2b, np, 64, 1);
    // gmp + lin1 -> xc[:, 128:]
    fill_f32<<<ce(NG * 64, T), T, 0, stream>>>(WS + pp, NG * 64, NEG);
    seg_max<<<ce((long long)np * 64, T), T, 0, stream>>>(ap2, bat_p, WS + pp, np, 64);
    gemm(WS + pp, l1W, l1b, WS + xc, NG, 64, 128, 256, 128, 1);

    // ================= head =================
    gemm(WS + xc, fc1W, fc1b, WS + t1, NG, 256, 1024, 1024, 0, 1);
    gemm(WS + t1, fc2W, fc2b, WS + t2, NG, 1024, 256, 256, 0, 1);
    head_out<<<ce(NG, T), T, 0, stream>>>(WS + t2, outW, outb, outp, NG, 256);
}
